// Net_14937896256213
// MI455X (gfx1250) — matrix operand order repaired and checked by reading, not compiled, not run on hardware
//
#include <hip/hip_runtime.h>
#include <hip/hip_bf16.h>

typedef _Float16 half_t;
typedef __attribute__((ext_vector_type(16))) _Float16 v16h;
typedef __attribute__((ext_vector_type(8)))  _Float16 v8h;
typedef __attribute__((ext_vector_type(4)))  _Float16 v4h;
typedef __attribute__((ext_vector_type(8)))  float    v8f;

// ---------------------------------------------------------------------------
// elementwise helpers
// ---------------------------------------------------------------------------
__global__ void fill_k(float* p, float v, long n) {
    long t = (long)blockIdx.x * blockDim.x + threadIdx.x;
    if (t < n) p[t] = v;
}

__global__ void bn_k(const float* __restrict__ in, float* __restrict__ out,
                     const float* __restrict__ g, const float* __restrict__ b,
                     const float* __restrict__ m, const float* __restrict__ v,
                     long n, int dim) {
    long t = (long)blockIdx.x * blockDim.x + threadIdx.x;
    if (t >= n) return;
    int c = (int)(t % dim);
    out[t] = (in[t] - m[c]) * rsqrtf(v[c] + 1e-5f) * g[c] + b[c];
}

// ---------------------------------------------------------------------------
// edge-parallel scatter:  agg[dst] += x[src] * em   (float4 lanes, f32 atomics)
// ---------------------------------------------------------------------------
__global__ void edge_agg_k(const float* __restrict__ x, const int* __restrict__ src,
                           const int* __restrict__ dst, const float* __restrict__ em,
                           float* __restrict__ agg, int E, int dim) {
    long t = (long)blockIdx.x * blockDim.x + threadIdx.x;
    int cpe = dim >> 2;
    long total = (long)E * cpe;
    if (t >= total) return;
    int e = (int)(t / cpe);
    int c = (int)(t % cpe) << 2;
    float m = em ? em[e] : 1.0f;
    if (m == 0.0f) return;
    const float4 v = *(const float4*)(x + (long)src[e] * dim + c);
    float* o = agg + (long)dst[e] * dim + c;
    atomicAdd(o + 0, v.x * m);
    atomicAdd(o + 1, v.y * m);
    atomicAdd(o + 2, v.z * m);
    atomicAdd(o + 3, v.w * m);
}

// ---------------------------------------------------------------------------
// WMMA GEMM:  C[M,N] = act( (A (+A2)) @ W[K,N] + bias )
// block = 128 threads (4 waves), tile 64x64, K-step 32 (f16 WMMA, f32 accum).
// B staged TRANSPOSED in LDS -> fragments are aligned ds_load_b128 pairs.
// Software-pipelined: next K-step's global tiles prefetched into registers
// while the current step's WMMAs execute. K and A2-presence templated.
// ---------------------------------------------------------------------------
#define KSTEP 32
#define LDAH  40   // padded row stride in halves (80B, 16B-aligned)

template <int K, bool HASA2>
__global__ __launch_bounds__(128) void gemm_wmma_k(
    const float* __restrict__ A, const float* __restrict__ A2,
    const float* __restrict__ W, const float* __restrict__ bias,
    float* __restrict__ C, int M, int N, int relu) {
    constexpr int NSTEPS = K / KSTEP;
    __shared__ alignas(16) half_t lA[64 * LDAH];    // [row m][k]
    __shared__ alignas(16) half_t lBt[64 * LDAH];   // [col n][k]  (transposed)
    const int bm = blockIdx.y * 64;
    const int bn = blockIdx.x * 64;
    const int tid  = threadIdx.x;
    const int wave = tid >> 5;
    const int lane = tid & 31;
    const int lr   = lane & 15;
    const bool hi  = lane >= 16;

    v8f acc0 = {}, acc1 = {}, acc2 = {}, acc3 = {};

    const int am = (wave << 4) + lr;   // A row this lane owns
    const int ka = hi ? 16 : 0;        // A frag: K {ka..ka+15}
    const int kb = hi ? 16 : 0;        // B frag: K {kb..kb+15}

    // staging registers (double-buffer against LDS)
    float4 ra[4], ra2[4], rb[4];

    // ---- prefetch K-step 0
#pragma unroll
    for (int i = 0; i < 4; ++i) {
        int e4 = tid + i * 128;                 // 512 float4 = 2048 elems
        int r = e4 >> 3, c4 = (e4 & 7) << 2;
        long gi = (long)(bm + r) * K + c4;
        ra[i] = *(const float4*)(A + gi);
        if (HASA2) ra2[i] = *(const float4*)(A2 + gi);
    }
#pragma unroll
    for (int i = 0; i < 4; ++i) {
        int e4 = tid + i * 128;
        int r = e4 >> 4, c4 = (e4 & 15) << 2;   // r = k row, c4 = n col
        rb[i] = *(const float4*)(W + (long)r * N + (bn + c4));
    }

#pragma unroll
    for (int kk = 0; kk < NSTEPS; ++kk) {
        // ---- commit staged tile to LDS (f32 -> f16)
#pragma unroll
        for (int i = 0; i < 4; ++i) {
            int e4 = tid + i * 128;
            int r = e4 >> 3, c4 = (e4 & 7) << 2;
            float4 v = ra[i];
            if (HASA2) {
                v.x += ra2[i].x; v.y += ra2[i].y; v.z += ra2[i].z; v.w += ra2[i].w;
            }
            v4h h4 = {(half_t)v.x, (half_t)v.y, (half_t)v.z, (half_t)v.w};
            *(v4h*)&lA[r * LDAH + c4] = h4;
        }
#pragma unroll
        for (int i = 0; i < 4; ++i) {
            int e4 = tid + i * 128;
            int r = e4 >> 4, c4 = (e4 & 15) << 2;
            float4 w = rb[i];
            lBt[(c4 + 0) * LDAH + r] = (half_t)w.x;
            lBt[(c4 + 1) * LDAH + r] = (half_t)w.y;
            lBt[(c4 + 2) * LDAH + r] = (half_t)w.z;
            lBt[(c4 + 3) * LDAH + r] = (half_t)w.w;
        }
        __syncthreads();

        // ---- prefetch next K-step while WMMAs run
        if (kk + 1 < NSTEPS) {
            const int k0n = (kk + 1) * KSTEP;
#pragma unroll
            for (int i = 0; i < 4; ++i) {
                int e4 = tid + i * 128;
                int r = e4 >> 3, c4 = (e4 & 7) << 2;
                long gi = (long)(bm + r) * K + (k0n + c4);
                ra[i] = *(const float4*)(A + gi);
                if (HASA2) ra2[i] = *(const float4*)(A2 + gi);
            }
#pragma unroll
            for (int i = 0; i < 4; ++i) {
                int e4 = tid + i * 128;
                int r = e4 >> 4, c4 = (e4 & 15) << 2;
                rb[i] = *(const float4*)(W + (long)(k0n + r) * N + (bn + c4));
            }
        }

        // ---- A fragment: two aligned b128 LDS loads
        const half_t* ap = &lA[am * LDAH + ka];
        v8h a0 = *(const v8h*)(ap);
        v8h a1 = *(const v8h*)(ap + 8);
        v16h af = __builtin_shufflevector(a0, a1, 0, 1, 2, 3, 4, 5, 6, 7,
                                          8, 9, 10, 11, 12, 13, 14, 15);

        // ---- B fragments: two aligned b128 LDS loads per n-tile
        {
            const half_t* bp = &lBt[(0 * 16 + lr) * LDAH + kb];
            v8h b0 = *(const v8h*)(bp);
            v8h b1 = *(const v8h*)(bp + 8);
            v16h bf = __builtin_shufflevector(b0, b1, 0, 1, 2, 3, 4, 5, 6, 7,
                                              8, 9, 10, 11, 12, 13, 14, 15);
            acc0 = __builtin_amdgcn_wmma_f32_16x16x32_f16(false, af, false, bf, (short)0, acc0, false, false);
        }
        {
            const half_t* bp = &lBt[(1 * 16 + lr) * LDAH + kb];
            v8h b0 = *(const v8h*)(bp);
            v8h b1 = *(const v8h*)(bp + 8);
            v16h bf = __builtin_shufflevector(b0, b1, 0, 1, 2, 3, 4, 5, 6, 7,
                                              8, 9, 10, 11, 12, 13, 14, 15);
            acc1 = __builtin_amdgcn_wmma_f32_16x16x32_f16(false, af, false, bf, (short)0, acc1, false, false);
        }
        {
            const half_t* bp = &lBt[(2 * 16 + lr) * LDAH + kb];
            v8h b0 = *(const v8h*)(bp);
            v8h b1 = *(const v8h*)(bp + 8);
            v16h bf = __builtin_shufflevector(b0, b1, 0, 1, 2, 3, 4, 5, 6, 7,
                                              8, 9, 10, 11, 12, 13, 14, 15);
            acc2 = __builtin_amdgcn_wmma_f32_16x16x32_f16(false, af, false, bf, (short)0, acc2, false, false);
        }
        {
            const half_t* bp = &lBt[(3 * 16 + lr) * LDAH + kb];
            v8h b0 = *(const v8h*)(bp);
            v8h b1 = *(const v8h*)(bp + 8);
            v16h bf = __builtin_shufflevector(b0, b1, 0, 1, 2, 3, 4, 5, 6, 7,
                                              8, 9, 10, 11, 12, 13, 14, 15);
            acc3 = __builtin_amdgcn_wmma_f32_16x16x32_f16(false, af, false, bf, (short)0, acc3, false, false);
        }
        __syncthreads();
    }

    // epilogue: C/D layout -> lanes 0-15 rows r, lanes 16-31 rows 8+r, col = lane&15
    const long rowbase = bm + (wave << 4) + (hi ? 8 : 0);
    const int  colb    = bn + lr;
#pragma unroll
    for (int r = 0; r < 8; ++r) {
        float* cp = C + (rowbase + r) * (long)N;
        float v;
        v = acc0[r] + bias[colb];      if (relu) v = fmaxf(v, 0.f); cp[colb]      = v;
        v = acc1[r] + bias[colb + 16]; if (relu) v = fmaxf(v, 0.f); cp[colb + 16] = v;
        v = acc2[r] + bias[colb + 32]; if (relu) v = fmaxf(v, 0.f); cp[colb + 32] = v;
        v = acc3[r] + bias[colb + 48]; if (relu) v = fmaxf(v, 0.f); cp[colb + 48] = v;
    }
}

// ---------------------------------------------------------------------------
// GCN score pieces
// ---------------------------------------------------------------------------
__global__ void node_dot_k(const float* __restrict__ x, const float* __restrict__ w,
                           float* __restrict__ h, int n, int dim) {
    int t = blockIdx.x * blockDim.x + threadIdx.x;
    if (t >= n) return;
    const float4* xp = (const float4*)(x + (long)t * dim);
    const float4* wp = (const float4*)w;
    float acc = 0.f;
    for (int i = 0; i < (dim >> 2); ++i) {
        float4 a = xp[i], b = wp[i];
        acc += a.x * b.x + a.y * b.y + a.z * b.z + a.w * b.w;
    }
    h[t] = acc;
}

__global__ void edge_deg_k(const int* __restrict__ dst, const float* __restrict__ em,
                           float* __restrict__ deg, int E) {
    int t = blockIdx.x * blockDim.x + threadIdx.x;
    if (t >= E) return;
    float m = em ? em[t] : 1.0f;
    if (m != 0.0f) atomicAdd(&deg[dst[t]], m);
}

__global__ void gcn_edge_k(const int* __restrict__ src, const int* __restrict__ dst,
                           const float* __restrict__ em, const float* __restrict__ h,
                           const float* __restrict__ deg, float* __restrict__ out, int E) {
    int t = blockIdx.x * blockDim.x + threadIdx.x;
    if (t >= E) return;
    float m = em ? em[t] : 1.0f;
    if (m == 0.0f) return;
    int s = src[t], d = dst[t];
    atomicAdd(&out[d], h[s] * rsqrtf(deg[s]) * rsqrtf(deg[d]) * m);
}

__global__ void score_fin_k(const float* sct, const float* h, const float* deg,
                            const float* b, float* score, int n) {
    int t = blockIdx.x * blockDim.x + threadIdx.x;
    if (t >= n) return;
    score[t] = sct[t] + h[t] / deg[t] + b[0];
}

// ---------------------------------------------------------------------------
// per-graph top-k by rank counting (matches lax.top_k order incl. tie-break)
// ---------------------------------------------------------------------------
__global__ void topk_k(const float* __restrict__ score, int npg, int k,
                       int* __restrict__ perm, int* __restrict__ o2n) {
    __shared__ float s[512];
    int g = blockIdx.x;
    for (int i = threadIdx.x; i < npg; i += blockDim.x) s[i] = score[g * npg + i];
    __syncthreads();
    for (int i = threadIdx.x; i < npg; i += blockDim.x) {
        float si = s[i];
        int rank = 0;
        for (int j = 0; j < npg; ++j) {
            float sj = s[j];
            rank += (sj > si) || (sj == si && j < i);
        }
        int old = g * npg + i;
        if (rank < k) {
            perm[g * k + rank] = old;
            o2n[old] = g * k + rank;
        } else {
            o2n[old] = -1;
        }
    }
}

__global__ void compact_k(const float* __restrict__ x, const float* __restrict__ score,
                          const int* __restrict__ perm, float* __restrict__ out,
                          int nk, int dim) {
    long t = (long)blockIdx.x * blockDim.x + threadIdx.x;
    int cpe = dim >> 2;
    long total = (long)nk * cpe;
    if (t >= total) return;
    int i = (int)(t / cpe);
    int c = (int)(t % cpe) << 2;
    int old = perm[i];
    float tn = tanhf(score[old]);
    float4 v = *(const float4*)(x + (long)old * dim + c);
    v.x *= tn; v.y *= tn; v.z *= tn; v.w *= tn;
    *(float4*)(out + (long)i * dim + c) = v;
}

__global__ void remap_k(const int* __restrict__ src, const int* __restrict__ dst,
                        const float* __restrict__ em, const int* __restrict__ o2n,
                        int* __restrict__ so, int* __restrict__ dd, float* __restrict__ mo,
                        int E) {
    int t = blockIdx.x * blockDim.x + threadIdx.x;
    if (t >= E) return;
    int ns = o2n[src[t]], nd = o2n[dst[t]];
    float m = (em ? em[t] : 1.0f) * ((ns >= 0 && nd >= 0) ? 1.0f : 0.0f);
    so[t] = ns < 0 ? 0 : ns;
    dd[t] = nd < 0 ? 0 : nd;
    mo[t] = m;
}

// ---------------------------------------------------------------------------
// unpool
// ---------------------------------------------------------------------------
__global__ void scatter_k(const float* __restrict__ xc, const int* __restrict__ perm,
                          float* __restrict__ base, int nc, int dim) {
    long t = (long)blockIdx.x * blockDim.x + threadIdx.x;
    int cpe = dim >> 2;
    long total = (long)nc * cpe;
    if (t >= total) return;
    int i = (int)(t / cpe);
    int c = (int)(t % cpe) << 2;
    *(float4*)(base + (long)perm[i] * dim + c) = *(const float4*)(xc + (long)i * dim + c);
}

__global__ void unpool_fin_k(const float* __restrict__ base, const float* __restrict__ neigh,
                             const float* __restrict__ deg, const int* __restrict__ o2n,
                             const float* __restrict__ skip, const float* __restrict__ res,
                             float* __restrict__ out, long n, int dim) {
    long t = (long)blockIdx.x * blockDim.x + threadIdx.x;
    if (t >= n) return;
    int i = (int)(t / dim);
    float v = (o2n[i] >= 0) ? base[t] : neigh[t] / fmaxf(deg[i], 1.0f);
    out[t] = v + res[0] * skip[t];
}

// ---------------------------------------------------------------------------
// readout: cat(max,mean) -> 2-way attention softmax -> reweight -> BN
// one block per graph, 256 threads (== DIM)
// ---------------------------------------------------------------------------
__global__ void readout_k(const float* __restrict__ x, int npg,
                          const float* __restrict__ aw, const float* __restrict__ ab,
                          const float* __restrict__ bg, const float* __restrict__ bb,
                          const float* __restrict__ bm, const float* __restrict__ bv,
                          float* __restrict__ out) {
    const int dim = 256;
    __shared__ float sg[512];
    __shared__ float r0[256], r1[256];
    __shared__ float a01[2];
    int g = blockIdx.x, c = threadIdx.x;
    float mx = -3.4e38f, sm = 0.f;
    const float* xp = x + (long)g * npg * dim + c;
    for (int r = 0; r < npg; ++r) {
        float v = xp[(long)r * dim];
        mx = fmaxf(mx, v);
        sm += v;
    }
    sg[c] = mx;
    sg[dim + c] = sm / npg;
    __syncthreads();
    r0[c] = sg[c] * aw[c * 2 + 0] + sg[dim + c] * aw[(dim + c) * 2 + 0];
    r1[c] = sg[c] * aw[c * 2 + 1] + sg[dim + c] * aw[(dim + c) * 2 + 1];
    __syncthreads();
    for (int s = 128; s > 0; s >>= 1) {
        if (c < s) { r0[c] += r0[c + s]; r1[c] += r1[c + s]; }
        __syncthreads();
    }
    if (c == 0) {
        float l0 = r0[0] + ab[0], l1 = r1[0] + ab[1];
        float m2 = fmaxf(l0, l1);
        float e0 = expf(l0 - m2), e1 = expf(l1 - m2);
        a01[0] = e0 / (e0 + e1);
        a01[1] = e1 / (e0 + e1);
    }
    __syncthreads();
    float v0 = sg[c] * a01[0];
    float v1 = sg[dim + c] * a01[1];
    out[(long)g * 512 + c]       = (v0 - bm[c]) * rsqrtf(bv[c] + 1e-5f) * bg[c] + bb[c];
    out[(long)g * 512 + dim + c] = (v1 - bm[dim + c]) * rsqrtf(bv[dim + c] + 1e-5f) * bg[dim + c] + bb[dim + c];
}

// ---------------------------------------------------------------------------
// final head: concat -> 3-way attention -> BN -> linear+relu -> log_softmax
// one block per graph, 256 threads
// ---------------------------------------------------------------------------
__global__ void head_k(const float* __restrict__ r2, const float* __restrict__ r3,
                       const float* __restrict__ r4,
                       const float* __restrict__ aw, const float* __restrict__ ab,
                       const float* __restrict__ bg, const float* __restrict__ bb,
                       const float* __restrict__ bm, const float* __restrict__ bv,
                       const float* __restrict__ lw, const float* __restrict__ lb,
                       float* __restrict__ out) {
    __shared__ float gl[1536];
    __shared__ float red[3][256];
    __shared__ float aa[3];
    __shared__ float logits[10];
    int g = blockIdx.x, t = threadIdx.x;
    for (int i = t; i < 512; i += 256) {
        gl[i]        = r2[(long)g * 512 + i];
        gl[512 + i]  = r3[(long)g * 512 + i];
        gl[1024 + i] = r4[(long)g * 512 + i];
    }
    __syncthreads();
    float p0 = 0.f, p1 = 0.f, p2 = 0.f;
    for (int i = t; i < 1536; i += 256) {
        float v = gl[i];
        p0 += v * aw[i * 3 + 0];
        p1 += v * aw[i * 3 + 1];
        p2 += v * aw[i * 3 + 2];
    }
    red[0][t] = p0; red[1][t] = p1; red[2][t] = p2;
    __syncthreads();
    for (int s = 128; s > 0; s >>= 1) {
        if (t < s) {
            red[0][t] += red[0][t + s];
            red[1][t] += red[1][t + s];
            red[2][t] += red[2][t + s];
        }
        __syncthreads();
    }
    if (t == 0) {
        float l0 = red[0][0] + ab[0], l1 = red[1][0] + ab[1], l2 = red[2][0] + ab[2];
        float m = fmaxf(l0, fmaxf(l1, l2));
        float e0 = expf(l0 - m), e1 = expf(l1 - m), e2 = expf(l2 - m);
        float s = e0 + e1 + e2;
        aa[0] = e0 / s; aa[1] = e1 / s; aa[2] = e2 / s;
    }
    __syncthreads();
    for (int i = t; i < 1536; i += 256) {
        float v = gl[i] * aa[i / 512];
        gl[i] = (v - bm[i]) * rsqrtf(bv[i] + 1e-5f) * bg[i] + bb[i];
    }
    __syncthreads();
    if (t < 10) {
        float acc = lb[t];
        for (int i = 0; i < 1536; ++i) acc += gl[i] * lw[i * 10 + t];
        logits[t] = fmaxf(acc, 0.f);
    }
    __syncthreads();
    if (t == 0) {
        float m = logits[0];
        for (int j = 1; j < 10; ++j) m = fmaxf(m, logits[j]);
        float s = 0.f;
        for (int j = 0; j < 10; ++j) s += expf(logits[j] - m);
        float lse = logf(s);
        for (int j = 0; j < 10; ++j) out[g * 10 + j] = logits[j] - m - lse;
    }
}

// ---------------------------------------------------------------------------
// host orchestration
// ---------------------------------------------------------------------------
extern "C" void kernel_launch(void* const* d_in, const int* in_sizes, int n_in,
                              void* d_out, int out_size, void* d_ws, size_t ws_size,
                              hipStream_t stream) {
    (void)in_sizes; (void)n_in; (void)out_size; (void)ws_size;
    constexpr int N0 = 32768, E = 524288, B = 64, DIM = 256, INCH = 128;
    constexpr int n1 = N0 / 2, n2 = N0 / 4;

    const float* X    = (const float*)d_in[0];
    const int*   ESRC = (const int*)d_in[1];
    const int*   EDST = (const int*)d_in[2];
    // flattened param order: sorted pytree keys
    const float* attn_b  = (const float*)d_in[3];
    const float* attn_w  = (const float*)d_in[4];
    const float* at1_b = (const float*)d_in[5],  *at1_w = (const float*)d_in[6];
    const float* at2_b = (const float*)d_in[7],  *at2_w = (const float*)d_in[8];
    const float* at3_b = (const float*)d_in[9],  *at3_w = (const float*)d_in[10];
    auto BN = [&](int base, const float*& g, const float*& b, const float*& m, const float*& v) {
        b = (const float*)d_in[base + 0]; g = (const float*)d_in[base + 1];
        m = (const float*)d_in[base + 2]; v = (const float*)d_in[base + 3];
    };
    const float *bn1g,*bn1b,*bn1m,*bn1v; BN(11, bn1g,bn1b,bn1m,bn1v);
    const float *bn2g,*bn2b,*bn2m,*bn2v; BN(15, bn2g,bn2b,bn2m,bn2v);
    const float *bn3g,*bn3b,*bn3m,*bn3v; BN(19, bn3g,bn3b,bn3m,bn3v);
    const float *bn4g,*bn4b,*bn4m,*bn4v; BN(23, bn4g,bn4b,bn4m,bn4v);
    const float *bn5g,*bn5b,*bn5m,*bn5v; BN(27, bn5g,bn5b,bn5m,bn5v);
    const float *bn6g,*bn6b,*bn6m,*bn6v; BN(31, bn6g,bn6b,bn6m,bn6v);
    const float *b61g,*b61b,*b61m,*b61v; BN(35, b61g,b61b,b61m,b61v);
    const float *b62g,*b62b,*b62m,*b62v; BN(39, b62g,b62b,b62m,b62v);
    const float *b63g,*b63b,*b63m,*b63v; BN(43, b63g,b63b,b63m,b63v);
    auto CONV = [&](int base, const float*& b1, const float*& b2, const float*& w1, const float*& w2) {
        b1 = (const float*)d_in[base + 0]; b2 = (const float*)d_in[base + 1];
        w1 = (const float*)d_in[base + 2]; w2 = (const float*)d_in[base + 3];
    };
    const float *c1b1,*c1b2,*c1w1,*c1w2; CONV(47, c1b1,c1b2,c1w1,c1w2);
    const float *c2b1,*c2b2,*c2w1,*c2w2; CONV(51, c2b1,c2b2,c2w1,c2w2);
    const float *c3b1,*c3b2,*c3w1,*c3w2; CONV(55, c3b1,c3b2,c3w1,c3w2);
    const float *c4b1,*c4b2,*c4w1,*c4w2; CONV(59, c4b1,c4b2,c4w1,c4w2);
    const float *c5b1,*c5b2,*c5w1,*c5w2; CONV(63, c5b1,c5b2,c5w1,c5w2);
    const float* lin_b = (const float*)d_in[67];
    const float* lin_w = (const float*)d_in[68];
    const float* p1b = (const float*)d_in[69], *p1w = (const float*)d_in[70];
    const float* p2b = (const float*)d_in[71], *p2w = (const float*)d_in[72];
    const float* res1 = (const float*)d_in[73];
    const float* res2 = (const float*)d_in[74];

    // workspace layout
    size_t off = 0;
    auto ALLOC = [&](size_t nbytes) -> void* {
        void* p = (char*)d_ws + off;
        off += (nbytes + 255) & ~(size_t)255;
        return p;
    };
    float* agg   = (float*)ALLOC((size_t)N0 * DIM * 4);  // also neigh
    float* hbuf  = (float*)ALLOC((size_t)N0 * DIM * 4);  // also base
    float* tmp   = (float*)ALLOC((size_t)N0 * DIM * 4);  // level input staging
    float* x0b   = (float*)ALLOC((size_t)N0 * DIM * 4);
    float* x4b   = (float*)ALLOC((size_t)N0 * DIM * 4);
    float* x1b   = (float*)ALLOC((size_t)n1 * DIM * 4);
    float* x3b   = (float*)ALLOC((size_t)n1 * DIM * 4);
    float* x2b   = (float*)ALLOC((size_t)n2 * DIM * 4);
    float* score = (float*)ALLOC((size_t)N0 * 4);
    float* hsc   = (float*)ALLOC((size_t)N0 * 4);
    float* deg   = (float*)ALLOC((size_t)N0 * 4);
    float* sct   = (float*)ALLOC((size_t)N0 * 4);
    int* o2n1 = (int*)ALLOC((size_t)N0 * 4);
    int* o2n2 = (int*)ALLOC((size_t)n1 * 4);
    int* perm1 = (int*)ALLOC((size_t)n1 * 4);
    int* perm2 = (int*)ALLOC((size_t)n2 * 4);
    int* s1 = (int*)ALLOC((size_t)E * 4);
    int* d1 = (int*)ALLOC((size_t)E * 4);
    float* m1 = (float*)ALLOC((size_t)E * 4);
    int* s2 = (int*)ALLOC((size_t)E * 4);
    int* d2 = (int*)ALLOC((size_t)E * 4);
    float* m2 = (float*)ALLOC((size_t)E * 4);
    float* r2 = (float*)ALLOC((size_t)B * 512 * 4);
    float* r3 = (float*)ALLOC((size_t)B * 512 * 4);
    float* r4 = (float*)ALLOC((size_t)B * 512 * 4);

    auto NB = [](long n) { return (unsigned)((n + 255) / 256); };
    auto zero = [&](float* p, long n) { fill_k<<<NB(n), 256, 0, stream>>>(p, 0.0f, n); };

    auto gemm = [&](const float* A, const float* A2, const float* W, const float* bias,
                    float* C, int M, int N, int K, int relu) {
        dim3 gr((unsigned)(N / 64), (unsigned)(M / 64));
        if (K == 128) {
            if (A2) gemm_wmma_k<128, true><<<gr, 128, 0, stream>>>(A, A2, W, bias, C, M, N, relu);
            else    gemm_wmma_k<128, false><<<gr, 128, 0, stream>>>(A, A2, W, bias, C, M, N, relu);
        } else {
            if (A2) gemm_wmma_k<256, true><<<gr, 128, 0, stream>>>(A, A2, W, bias, C, M, N, relu);
            else    gemm_wmma_k<256, false><<<gr, 128, 0, stream>>>(A, A2, W, bias, C, M, N, relu);
        }
    };

    // GIN: out = act2( relu((x+agg)@w1+b1) @ w2 + b2 )
    auto gin = [&](const float* xin, const int* src, const int* dst, const float* em,
                   int n, int dimIn, const float* w1, const float* b1,
                   const float* w2, const float* b2, float* outp, int outer_relu) {
        zero(agg, (long)n * dimIn);
        edge_agg_k<<<NB((long)E * (dimIn >> 2)), 256, 0, stream>>>(xin, src, dst, em, agg, E, dimIn);
        gemm(xin, agg, w1, b1, hbuf, n, DIM, dimIn, 1);
        gemm(hbuf, nullptr, w2, b2, outp, n, DIM, DIM, outer_relu);
    };

    // SAGPool
    auto sag = [&](const float* xin, const int* src, const int* dst, const float* em,
                   int n, int npg, const float* pw, const float* pb,
                   int* o2n, int* perm, float* xout, int* so, int* dd, float* mo) {
        int k = npg / 2, nk = n / 2;
        node_dot_k<<<NB(n), 256, 0, stream>>>(xin, pw, hsc, n, DIM);
        fill_k<<<NB(n), 256, 0, stream>>>(deg, 1.0f, n);
        edge_deg_k<<<NB(E), 256, 0, stream>>>(dst, em, deg, E);
        zero(sct, n);
        gcn_edge_k<<<NB(E), 256, 0, stream>>>(src, dst, em, hsc, deg, sct, E);
        score_fin_k<<<NB(n), 256, 0, stream>>>(sct, hsc, deg, pb, score, n);
        topk_k<<<B, 256, 0, stream>>>(score, npg, k, perm, o2n);
        compact_k<<<NB((long)nk * (DIM >> 2)), 256, 0, stream>>>(xin, score, perm, xout, nk, DIM);
        remap_k<<<NB(E), 256, 0, stream>>>(src, dst, em, o2n, so, dd, mo, E);
    };

    // unpool + residual into outp
    auto unpool = [&](const float* xc, int nc, const int* src, const int* dst,
                      const float* em, const int* perm, const int* o2n,
                      const float* skip, const float* res, int nf, float* outp) {
        float* base = hbuf;
        float* neigh = agg;
        zero(base, (long)nf * DIM);
        scatter_k<<<NB((long)nc * (DIM >> 2)), 256, 0, stream>>>(xc, perm, base, nc, DIM);
        zero(deg, nf);
        edge_deg_k<<<NB(E), 256, 0, stream>>>(dst, em, deg, E);
        zero(neigh, (long)nf * DIM);
        edge_agg_k<<<NB((long)E * (DIM >> 2)), 256, 0, stream>>>(base, src, dst, em, neigh, E, DIM);
        unpool_fin_k<<<NB((long)nf * DIM), 256, 0, stream>>>(base, neigh, deg, o2n, skip, res, outp,
                                                             (long)nf * DIM, DIM);
    };

    // ---- level 0: conv1 -> x0 ----
    gin(X, ESRC, EDST, nullptr, N0, INCH, c1w1, c1b1, c1w2, c1b2, x0b, 1);

    // ---- pool1 -> tmp(x1raw), edges (s1,d1,m1) ----
    sag(x0b, ESRC, EDST, nullptr, N0, 512, p1w, p1b, o2n1, perm1, tmp, s1, d1, m1);
    bn_k<<<NB((long)n1 * DIM), 256, 0, stream>>>(tmp, tmp, bn1g, bn1b, bn1m, bn1v, (long)n1 * DIM, DIM);

    // ---- conv2 -> x1 ----
    gin(tmp, s1, d1, m1, n1, DIM, c2w1, c2b1, c2w2, c2b2, x1b, 1);

    // ---- pool2 -> tmp(x2raw), edges (s2,d2,m2) ----
    sag(x1b, s1, d1, m1, n1, 256, p2w, p2b, o2n2, perm2, tmp, s2, d2, m2);
    bn_k<<<NB((long)n2 * DIM), 256, 0, stream>>>(tmp, tmp, bn2g, bn2b, bn2m, bn2v, (long)n2 * DIM, DIM);

    // ---- conv3 -> x2 (then bn3) ----
    gin(tmp, s2, d2, m2, n2, DIM, c3w1, c3b1, c3w2, c3b2, x2b, 1);
    bn_k<<<NB((long)n2 * DIM), 256, 0, stream>>>(x2b, x2b, bn3g, bn3b, bn3m, bn3v, (long)n2 * DIM, DIM);

    // ---- unpool 2->1, conv4 -> x3 (then bn4) ----
    unpool(x2b, n2, s1, d1, m1, perm2, o2n2, x1b, res1, n1, tmp);
    gin(tmp, s1, d1, m1, n1, DIM, c4w1, c4b1, c4w2, c4b2, x3b, 1);
    bn_k<<<NB((long)n1 * DIM), 256, 0, stream>>>(x3b, x3b, bn4g, bn4b, bn4m, bn4v, (long)n1 * DIM, DIM);

    // ---- unpool 1->0, conv5 -> x4 (then bn5) ----
    unpool(x3b, n1, ESRC, EDST, nullptr, perm1, o2n1, x0b, res2, N0, tmp);
    gin(tmp, ESRC, EDST, nullptr, N0, DIM, c5w1, c5b1, c5w2, c5b2, x4b, 1);
    bn_k<<<NB((long)N0 * DIM), 256, 0, stream>>>(x4b, x4b, bn5g, bn5b, bn5m, bn5v, (long)N0 * DIM, DIM);

    // ---- readouts ----
    readout_k<<<B, 256, 0, stream>>>(x2b, 128, at1_w, at1_b, b61g, b61b, b61m, b61v, r2);
    readout_k<<<B, 256, 0, stream>>>(x3b, 256, at2_w, at2_b, b62g, b62b, b62m, b62v, r3);
    readout_k<<<B, 256, 0, stream>>>(x4b, 512, at3_w, at3_b, b63g, b63b, b63m, b63v, r4);

    // ---- final head -> log-softmax logits [64,10] ----
    head_k<<<B, 256, 0, stream>>>(r2, r3, r4, attn_w, attn_b, bn6g, bn6b, bn6m, bn6v,
                                  lin_w, lin_b, (float*)d_out);
}
